// Searcher_16793322127534
// MI455X (gfx1250) — matrix operand order repaired and checked by reading, not compiled, not run on hardware
//
#include <hip/hip_runtime.h>

// ---------------------------------------------------------------------------
// CDNA5 (gfx1250) WMMA types
// ---------------------------------------------------------------------------
typedef __attribute__((ext_vector_type(16))) __bf16 v16bf;
typedef __attribute__((ext_vector_type(8)))  float  v8f;

__device__ __forceinline__ float sigm(float x) { return 1.f / (1.f + expf(-x)); }

// D = 512, V = 32000, SL = 64, BS = 16, BW = 5, N = 80, G = 2048, T_DEC = 96
#define TD_OUT 7680   // 96*16*5 elements per output section

__device__ __forceinline__ v16bf load_afrag(const __bf16* p) {
    // A layout (16-bit 16x32): lane<16 -> K = {0..7, 16..23}; lane>=16 -> +8
    v16bf a;
#pragma unroll
    for (int i = 0; i < 8; ++i) { a[i] = p[i]; a[i + 8] = p[i + 16]; }
    return a;
}
__device__ __forceinline__ v16bf load_bfrag(const __bf16* p) {
    // B layout (16-bit 32x16): lane<16 -> K = {0..7, 16..23}; lane>=16 -> +8
    v16bf b;
#pragma unroll
    for (int i = 0; i < 8; ++i) { b[i] = p[i]; b[i + 8] = p[i + 16]; }
    return b;
}

// ---------------------------------------------------------------------------
// Elementwise helpers
// ---------------------------------------------------------------------------
__global__ void k_cvt(const float* __restrict__ a, __bf16* __restrict__ o, long n) {
    long i = (long)blockIdx.x * 256 + threadIdx.x;
    if (i < n) o[i] = (__bf16)a[i];
}

__global__ void k_transpose_bf16(const float* __restrict__ a, __bf16* __restrict__ o,
                                 int rows, int cols) {
    long i = (long)blockIdx.x * 256 + threadIdx.x;
    if (i >= (long)rows * cols) return;
    int r = (int)(i / cols), c = (int)(i - (long)r * cols);
    o[(size_t)c * rows + r] = (__bf16)a[i];
}

// o[r][0:ka] = A[r], o[r][ka:ka+kb] = B[r]
__global__ void k_concat2(const float* __restrict__ A, int ka,
                          const float* __restrict__ B, int kb,
                          __bf16* __restrict__ o, int rows) {
    int kt = ka + kb;
    long i = (long)blockIdx.x * 256 + threadIdx.x;
    if (i >= (long)rows * kt) return;
    int r = (int)(i / kt), c = (int)(i - (long)r * kt);
    float v = (c < ka) ? A[(size_t)r * ka + c] : B[(size_t)r * kb + (c - ka)];
    o[i] = (__bf16)v;
}

__global__ void k_embed(const int* __restrict__ ids, const float* __restrict__ emb,
                        __bf16* __restrict__ o, int rows, int d) {
    long i = (long)blockIdx.x * 256 + threadIdx.x;
    if (i >= (long)rows * d) return;
    int r = (int)(i / d), c = (int)(i - (long)r * d);
    o[i] = (__bf16)emb[(size_t)ids[r] * d + c];
}

// ---------------------------------------------------------------------------
// Generic WMMA GEMM:  C[M,N] = act(A[M,K] * W[N,K]^T + bias[N])
// one wave (32 threads) per 16x64 output strip; grid = (N/64, M/16).
// Software-pipelined: fragments for k+32 are loaded while the four WMMAs for
// k execute, so L2 latency overlaps the matrix pipe.
// act: 0 = none, 1 = tanh.  Cbf (optional) gets a bf16 copy of C.
// ---------------------------------------------------------------------------
__global__ void __launch_bounds__(32)
k_gemm(const __bf16* __restrict__ A, int lda,
       const __bf16* __restrict__ W, int ldw,
       const float* __restrict__ bias,
       float* __restrict__ C, __bf16* __restrict__ Cbf,
       int ldc, int K, int act)
{
    int lane = threadIdx.x & 31;
    int half = lane >> 4;
    int r15  = lane & 15;
    int m0 = blockIdx.y * 16;
    int n0 = blockIdx.x * 64;

    v8f acc[4];
#pragma unroll
    for (int j = 0; j < 4; ++j) {
        float bv = bias ? bias[n0 + j * 16 + r15] : 0.f;
#pragma unroll
        for (int r = 0; r < 8; ++r) acc[j][r] = bv;
    }

    const __bf16* arow = A + (size_t)(m0 + r15) * lda + half * 8;
    const __bf16* wrow[4];
#pragma unroll
    for (int j = 0; j < 4; ++j)
        wrow[j] = W + (size_t)(n0 + j * 16 + r15) * ldw + half * 8;

    // prologue: fragments for k = 0
    v16bf a_cur = load_afrag(arow);
    v16bf b_cur0 = load_bfrag(wrow[0]);
    v16bf b_cur1 = load_bfrag(wrow[1]);
    v16bf b_cur2 = load_bfrag(wrow[2]);
    v16bf b_cur3 = load_bfrag(wrow[3]);

    for (int k = 0; k < K; k += 32) {
        v16bf a_nxt, b_nxt0, b_nxt1, b_nxt2, b_nxt3;
        int kn = k + 32;
        if (kn < K) {                       // prefetch next k-slice
            a_nxt  = load_afrag(arow + kn);
            b_nxt0 = load_bfrag(wrow[0] + kn);
            b_nxt1 = load_bfrag(wrow[1] + kn);
            b_nxt2 = load_bfrag(wrow[2] + kn);
            b_nxt3 = load_bfrag(wrow[3] + kn);
        }
        acc[0] = __builtin_amdgcn_wmma_f32_16x16x32_bf16(false, a_cur, false, b_cur0,
                                                         (short)0, acc[0], false, false);
        acc[1] = __builtin_amdgcn_wmma_f32_16x16x32_bf16(false, a_cur, false, b_cur1,
                                                         (short)0, acc[1], false, false);
        acc[2] = __builtin_amdgcn_wmma_f32_16x16x32_bf16(false, a_cur, false, b_cur2,
                                                         (short)0, acc[2], false, false);
        acc[3] = __builtin_amdgcn_wmma_f32_16x16x32_bf16(false, a_cur, false, b_cur3,
                                                         (short)0, acc[3], false, false);
        a_cur = a_nxt;
        b_cur0 = b_nxt0; b_cur1 = b_nxt1; b_cur2 = b_nxt2; b_cur3 = b_nxt3;
    }

#pragma unroll
    for (int j = 0; j < 4; ++j) {
        int n = n0 + j * 16 + r15;
#pragma unroll
        for (int r = 0; r < 8; ++r) {
            int m = m0 + r + half * 8;
            float v = acc[j][r];
            if (act == 1) v = tanhf(v);
            C[(size_t)m * ldc + n] = v;
            if (Cbf) Cbf[(size_t)m * ldc + n] = (__bf16)v;
        }
    }
}

// ---------------------------------------------------------------------------
// Persistent encoder LSTM scan (one direction per block; grid = 2).
// Recurrent GEMM: h[16,512] x Whh[2048,512]^T via WMMA; input-side x@Wih^T+b
// is precomputed in XW.  32 waves; wave w owns d-slice [16w,16w+16) and the
// 4 gate tiles for it; cell state c lives in registers across all 64 steps.
// B (Whh) fragments are prefetched one k-slice ahead to hide L2 latency.
// ---------------------------------------------------------------------------
__global__ void __launch_bounds__(1024)
k_enc_scan(const __bf16* __restrict__ Whh,   // [2][2048][512] bf16
           const float*  __restrict__ XW,    // [2][1024][2048] f32 (bias folded)
           float* __restrict__ outf,         // [64][16][1024], writes dir half
           __bf16* __restrict__ outb)        // nullable, same layout
{
    int dir = blockIdx.x;
    const __bf16* Wd = Whh + (size_t)dir * 2048 * 512;
    const float*  Xd = XW  + (size_t)dir * 1024 * 2048;
    int tid  = threadIdx.x;
    int lane = tid & 31;
    int wave = tid >> 5;
    int half = lane >> 4;
    int r15  = lane & 15;
    int d0   = wave * 16;

    __shared__ __bf16 h_lds[16][512];
    for (int i = tid; i < 16 * 512; i += 1024) (&h_lds[0][0])[i] = (__bf16)0.f;

    const __bf16* wrow[4];
#pragma unroll
    for (int g = 0; g < 4; ++g)
        wrow[g] = Wd + (size_t)(g * 512 + d0 + r15) * 512 + half * 8;

    v8f c_reg;
#pragma unroll
    for (int r = 0; r < 8; ++r) c_reg[r] = 0.f;
    __syncthreads();

    for (int s = 0; s < 64; ++s) {
        int t = dir ? (63 - s) : s;
        v8f acc[4];
#pragma unroll
        for (int g = 0; g < 4; ++g) {
            int n = g * 512 + d0 + r15;
#pragma unroll
            for (int r = 0; r < 8; ++r) {
                int m = r + half * 8;
                acc[g][r] = Xd[((size_t)t * 16 + m) * 2048 + n];
            }
        }
        // prologue for k = 0
        v16bf b0 = load_bfrag(wrow[0]);
        v16bf b1 = load_bfrag(wrow[1]);
        v16bf b2 = load_bfrag(wrow[2]);
        v16bf b3 = load_bfrag(wrow[3]);
        for (int k = 0; k < 512; k += 32) {
            v16bf a = load_afrag(&h_lds[r15][k + half * 8]);
            v16bf n0, n1, n2, n3;
            int kn = k + 32;
            if (kn < 512) {
                n0 = load_bfrag(wrow[0] + kn);
                n1 = load_bfrag(wrow[1] + kn);
                n2 = load_bfrag(wrow[2] + kn);
                n3 = load_bfrag(wrow[3] + kn);
            }
            acc[0] = __builtin_amdgcn_wmma_f32_16x16x32_bf16(false, a, false, b0,
                                                             (short)0, acc[0], false, false);
            acc[1] = __builtin_amdgcn_wmma_f32_16x16x32_bf16(false, a, false, b1,
                                                             (short)0, acc[1], false, false);
            acc[2] = __builtin_amdgcn_wmma_f32_16x16x32_bf16(false, a, false, b2,
                                                             (short)0, acc[2], false, false);
            acc[3] = __builtin_amdgcn_wmma_f32_16x16x32_bf16(false, a, false, b3,
                                                             (short)0, acc[3], false, false);
            b0 = n0; b1 = n1; b2 = n2; b3 = n3;
        }
        __syncthreads();   // all waves done reading h_lds
#pragma unroll
        for (int r = 0; r < 8; ++r) {
            int m = r + half * 8;
            int d = d0 + r15;
            float gi = acc[0][r], gf = acc[1][r], gg = acc[2][r], go = acc[3][r];
            float cn = sigm(gf) * c_reg[r] + sigm(gi) * tanhf(gg);
            float hn = sigm(go) * tanhf(cn);
            c_reg[r] = cn;
            h_lds[m][d] = (__bf16)hn;
            size_t o = ((size_t)t * 16 + m) * 1024 + (size_t)dir * 512 + d;
            outf[o] = hn;
            if (outb) outb[o] = (__bf16)hn;
        }
        __syncthreads();
    }
}

// ---------------------------------------------------------------------------
// Decoder helpers
// ---------------------------------------------------------------------------
__global__ void k_dec_init(float* att, float* h0, float* c0, float* h1, float* c1,
                           int* token, float* score, int* fin) {
    int i = blockIdx.x * 256 + threadIdx.x;
    if (i < 80 * 512) { att[i] = 0.f; h0[i] = 0.f; c0[i] = 0.f; h1[i] = 0.f; c1[i] = 0.f; }
    if (i < 80) {
        int w = i % 5;
        token[i] = w ? 0 : 1;          // SOS on beam 0
        score[i] = w ? -1e9f : 0.f;
        fin[i]   = w ? 1 : 0;
    }
}

// xcat0[n] = [tgt_emb[token[n]](512) | att(512) | h0(512)];  xcat1[n][512:] = h1
__global__ void k_build_in(const int* __restrict__ token, const float* __restrict__ tgt_emb,
                           const float* __restrict__ att_s, const float* __restrict__ h0_s,
                           const float* __restrict__ h1_s,
                           __bf16* __restrict__ xcat0, __bf16* __restrict__ xcat1) {
    int i = blockIdx.x * 256 + threadIdx.x;
    if (i >= 80 * 512) return;
    int n = i >> 9, d = i & 511;
    xcat0[(size_t)n * 1536 + d]        = (__bf16)tgt_emb[(size_t)token[n] * 512 + d];
    xcat0[(size_t)n * 1536 + 512 + d]  = (__bf16)att_s[i];
    xcat0[(size_t)n * 1536 + 1024 + d] = (__bf16)h0_s[i];
    xcat1[(size_t)n * 1024 + 512 + d]  = (__bf16)h1_s[i];
}

__global__ void k_lstm_pw(const float* __restrict__ Gm, const float* __restrict__ c_in,
                          float* __restrict__ c_out, float* __restrict__ h_out,
                          __bf16* __restrict__ hb, int hb_stride, int hb_off) {
    int i = blockIdx.x * 256 + threadIdx.x;
    if (i >= 80 * 512) return;
    int n = i >> 9, d = i & 511;
    float gi = Gm[(size_t)n * 2048 + d];
    float gf = Gm[(size_t)n * 2048 + 512 + d];
    float gg = Gm[(size_t)n * 2048 + 1024 + d];
    float go = Gm[(size_t)n * 2048 + 1536 + d];
    float cn = sigm(gf) * c_in[i] + sigm(gi) * tanhf(gg);
    float hn = sigm(go) * tanhf(cn);
    c_out[i] = cn; h_out[i] = hn;
    hb[(size_t)n * hb_stride + hb_off + d] = (__bf16)hn;
}

// attention: e = q . ann (masked), softmax, argmax, ctx -> cat_att[:, :1024]
__global__ void k_attn(const float* __restrict__ q,      // [80][1024]
                       const float* __restrict__ ann,    // [64][16][1024]
                       const int* __restrict__ enc_ids,  // [64][16]
                       __bf16* __restrict__ cat_att,     // [80][1536]
                       int* __restrict__ most_al) {
    __shared__ float qs[1024];
    __shared__ float part[256];
    __shared__ float e_s[64];
    __shared__ float al_s[64];
    __shared__ float inv_s;
    int n = blockIdx.x, b = n / 5, tid = threadIdx.x;
    for (int i = tid; i < 1024; i += 256) qs[i] = q[(size_t)n * 1024 + i];
    __syncthreads();
    int s = tid >> 2, pp = tid & 3;
    const float* arow = ann + ((size_t)s * 16 + b) * 1024;
    float acc = 0.f;
    for (int d = pp * 256; d < pp * 256 + 256; ++d) acc += qs[d] * arow[d];
    part[tid] = acc;
    __syncthreads();
    if (pp == 0) {
        float e = part[tid] + part[tid + 1] + part[tid + 2] + part[tid + 3];
        if (enc_ids[s * 16 + b] == 0) e += -1e9f;   // PAD mask
        e_s[s] = e;
    }
    __syncthreads();
    if (tid == 0) {
        float mx = e_s[0]; int am = 0;
        for (int i = 1; i < 64; ++i) if (e_s[i] > mx) { mx = e_s[i]; am = i; }
        float sum = 0.f;
        for (int i = 0; i < 64; ++i) { float v = expf(e_s[i] - mx); al_s[i] = v; sum += v; }
        inv_s = 1.f / sum;
        most_al[n] = am;
    }
    __syncthreads();
    float inv = inv_s;
    for (int d = tid; d < 1024; d += 256) {
        float cx = 0.f;
        for (int i = 0; i < 64; ++i)
            cx += al_s[i] * inv * ann[((size_t)i * 16 + b) * 1024 + d];
        cat_att[(size_t)n * 1536 + d] = (__bf16)cx;
    }
}

__global__ void k_lse(const float* __restrict__ logits, float* __restrict__ rowlse) {
    __shared__ float red[256];
    int n = blockIdx.x, tid = threadIdx.x;
    const float* row = logits + (size_t)n * 32000;
    float mx = -3.0e38f;
    for (int v = tid; v < 32000; v += 256) mx = fmaxf(mx, row[v]);
    red[tid] = mx; __syncthreads();
    for (int off = 128; off >= 1; off >>= 1) {
        if (tid < off) red[tid] = fmaxf(red[tid], red[tid + off]);
        __syncthreads();
    }
    mx = red[0]; __syncthreads();
    float ss = 0.f;
    for (int v = tid; v < 32000; v += 256) ss += expf(row[v] - mx);
    red[tid] = ss; __syncthreads();
    for (int off = 128; off >= 1; off >>= 1) {
        if (tid < off) red[tid] += red[tid + off];
        __syncthreads();
    }
    if (tid == 0) rowlse[n] = mx + logf(red[0]);
}

// per-batch top-5 over BW*V with log-softmax + fin penalty + running score
__global__ void k_topk(const float* __restrict__ logits, const float* __restrict__ rowlse,
                       float* score, int* token, int* fin, int* beam_buf,
                       const int* __restrict__ most_al, float* __restrict__ out, int t) {
    __shared__ float vs[256 * 5];
    __shared__ int   ix[256 * 5];
    __shared__ float fpen[5];
    __shared__ float sbase[5];
    int b = blockIdx.x, tid = threadIdx.x;
    if (tid < 5) {
        int n = b * 5 + tid;
        fpen[tid]  = fin[n] ? -1e9f : 0.f;
        sbase[tid] = score[n];
    }
    __syncthreads();
    float lv[5]; int li[5];
#pragma unroll
    for (int i = 0; i < 5; ++i) { lv[i] = -3.0e38f; li[i] = 0; }
    for (int j = tid; j < 5 * 32000; j += 256) {
        int w = j / 32000;
        int v = j - w * 32000;
        int n = b * 5 + w;
        float sc = (logits[(size_t)n * 32000 + v] - rowlse[n]) + fpen[w] + sbase[w];
        if (sc > lv[4]) {
            int p = 4;
            while (p > 0 && sc > lv[p - 1]) { lv[p] = lv[p - 1]; li[p] = li[p - 1]; --p; }
            lv[p] = sc; li[p] = j;
        }
    }
#pragma unroll
    for (int i = 0; i < 5; ++i) { vs[tid * 5 + i] = lv[i]; ix[tid * 5 + i] = li[i]; }
    for (int off = 128; off >= 1; off >>= 1) {
        __syncthreads();
        float mv[5] = {0, 0, 0, 0, 0}; int mi[5] = {0, 0, 0, 0, 0};
        if (tid < off) {
            int ia = 0, ib = 0;
#pragma unroll
            for (int k2 = 0; k2 < 5; ++k2) {
                float va  = vs[tid * 5 + ia];
                float vb2 = vs[(tid + off) * 5 + ib];
                if (va >= vb2) { mv[k2] = va;  mi[k2] = ix[tid * 5 + ia]; ++ia; }
                else           { mv[k2] = vb2; mi[k2] = ix[(tid + off) * 5 + ib]; ++ib; }
            }
        }
        __syncthreads();
        if (tid < off) {
#pragma unroll
            for (int k2 = 0; k2 < 5; ++k2) { vs[tid * 5 + k2] = mv[k2]; ix[tid * 5 + k2] = mi[k2]; }
        }
    }
    __syncthreads();
    if (tid == 0) {
        int fold[5];
#pragma unroll
        for (int w = 0; w < 5; ++w) fold[w] = fin[b * 5 + w];
        for (int w = 0; w < 5; ++w) {
            float val = vs[w];
            int j = ix[w];
            int tok = j % 32000;
            int bm  = j / 32000;
            int n = b * 5 + w;
            out[             t * 80 + n] = (float)tok;                     // toks
            out[    TD_OUT + t * 80 + n] = (float)bm;                      // beams
            out[2 * TD_OUT + t * 80 + n] = val;                            // scores
            out[3 * TD_OUT + t * 80 + n] = (float)most_al[b * 5 + bm];     // aligns
            token[n] = tok;
            score[n] = val;
            fin[n] = ((fold[bm] != 0) || (tok == 2)) ? 1 : 0;              // EOS=2
            beam_buf[n] = bm;
        }
    }
}

__global__ void k_gather(const int* __restrict__ beam,
                         const float* __restrict__ natt, const float* __restrict__ h0t,
                         const float* __restrict__ c0t,  const float* __restrict__ h1t,
                         const float* __restrict__ c1t,
                         float* att_s, float* h0_s, float* c0_s, float* h1_s, float* c1_s) {
    int i = blockIdx.x * 256 + threadIdx.x;
    if (i >= 80 * 512) return;
    int n = i >> 9, d = i & 511;
    int b = n / 5;
    size_t j = ((size_t)(b * 5 + beam[n])) * 512 + d;
    att_s[i] = natt[j]; h0_s[i] = h0t[j]; c0_s[i] = c0t[j];
    h1_s[i] = h1t[j];   c1_s[i] = c1t[j];
}

// ---------------------------------------------------------------------------
// Host driver
// ---------------------------------------------------------------------------
extern "C" void kernel_launch(void* const* d_in, const int* in_sizes, int n_in,
                              void* d_out, int out_size, void* d_ws, size_t ws_size,
                              hipStream_t stream)
{
    const int*   input_enc = (const int*)  d_in[0];
    const float* src_emb   = (const float*)d_in[1];
    const float* tgt_emb   = (const float*)d_in[2];
    const float* eWih0 = (const float*)d_in[3];
    const float* eWhh0 = (const float*)d_in[4];
    const float* eb0   = (const float*)d_in[5];
    const float* eWih1 = (const float*)d_in[6];
    const float* eWhh1 = (const float*)d_in[7];
    const float* eb1   = (const float*)d_in[8];
    const float* dWih0 = (const float*)d_in[9];
    const float* dWhh0 = (const float*)d_in[10];
    const float* db0   = (const float*)d_in[11];
    const float* dWih1 = (const float*)d_in[12];
    const float* dWhh1 = (const float*)d_in[13];
    const float* db1   = (const float*)d_in[14];
    const float* aWa   = (const float*)d_in[15];
    const float* aWc   = (const float*)d_in[16];
    const float* pW    = (const float*)d_in[17];
    const float* pb    = (const float*)d_in[18];
    float* out = (float*)d_out;

    char* p = (char*)d_ws;
    auto ab = [&](size_t e) { __bf16* r = (__bf16*)p; p += ((e * 2 + 255) & ~(size_t)255); return r; };
    auto af = [&](size_t e) { float*  r = (float*) p; p += ((e * 4 + 255) & ~(size_t)255); return r; };
    auto ai = [&](size_t e) { int*    r = (int*)   p; p += ((e * 4 + 255) & ~(size_t)255); return r; };

    // --- workspace layout ---
    __bf16* srcX    = ab(1024 * 512);
    __bf16* eWih0b  = ab((size_t)2 * 2048 * 512);
    __bf16* eWhh0b  = ab((size_t)2 * 2048 * 512);
    __bf16* eWih1b  = ab((size_t)2 * 2048 * 1024);
    __bf16* eWhh1b  = ab((size_t)2 * 2048 * 512);
    __bf16* x1b     = ab(1024 * 1024);
    __bf16* Wcat0b  = ab((size_t)2048 * 1536);
    __bf16* Wcat1b  = ab((size_t)2048 * 1024);
    __bf16* WaTb    = ab(1024 * 512);
    __bf16* Wcb     = ab(512 * 1536);
    __bf16* projWb  = ab((size_t)32000 * 512);
    __bf16* xcat0b  = ab(80 * 1536);
    __bf16* xcat1b  = ab(80 * 1024);
    __bf16* catab   = ab(80 * 1536);
    __bf16* nattb   = ab(80 * 512);

    float* XW    = af((size_t)2 * 1024 * 2048);
    float* x1f   = af(1024 * 1024);
    float* annf  = af(1024 * 1024);
    float* G0    = af(80 * 2048);
    float* G1    = af(80 * 2048);
    float* qbuf  = af(80 * 1024);
    float* logits= af((size_t)80 * 32000);
    float* att_s = af(80 * 512);
    float* h0_s  = af(80 * 512);
    float* c0_s  = af(80 * 512);
    float* h1_s  = af(80 * 512);
    float* c1_s  = af(80 * 512);
    float* h0t   = af(80 * 512);
    float* c0t   = af(80 * 512);
    float* h1t   = af(80 * 512);
    float* c1t   = af(80 * 512);
    float* nattf = af(80 * 512);
    float* score_s = af(80);
    float* rowlse  = af(80);
    int* token_s = ai(80);
    int* fin_s   = ai(80);
    int* beam_b  = ai(80);
    int* mosta   = ai(80);

    auto cvt = [&](const float* a, __bf16* o, long n) {
        k_cvt<<<(int)((n + 255) / 256), 256, 0, stream>>>(a, o, n);
    };
    auto gemm = [&](const __bf16* A, int lda, const __bf16* W, int ldw, const float* bias,
                    float* C, __bf16* Cbf, int ldc, int M, int Nn, int K, int act) {
        dim3 gr(Nn / 64, M / 16);
        k_gemm<<<gr, 32, 0, stream>>>(A, lda, W, ldw, bias, C, Cbf, ldc, K, act);
    };

    // --- weight prep (bf16, transposes, concatenations) ---
    cvt(eWih0, eWih0b, (long)2 * 2048 * 512);
    cvt(eWhh0, eWhh0b, (long)2 * 2048 * 512);
    cvt(eWih1, eWih1b, (long)2 * 2048 * 1024);
    cvt(eWhh1, eWhh1b, (long)2 * 2048 * 512);
    cvt(aWc,   Wcb,    (long)512 * 1536);
    cvt(pW,    projWb, (long)32000 * 512);
    k_transpose_bf16<<<(512 * 1024 + 255) / 256, 256, 0, stream>>>(aWa, WaTb, 512, 1024);
    k_concat2<<<((2048 * 1536) + 255) / 256, 256, 0, stream>>>(dWih0, 1024, dWhh0, 512, Wcat0b, 2048);
    k_concat2<<<((2048 * 1024) + 255) / 256, 256, 0, stream>>>(dWih1, 512, dWhh1, 512, Wcat1b, 2048);

    // --- encoder ---
    k_embed<<<(1024 * 512 + 255) / 256, 256, 0, stream>>>(input_enc, src_emb, srcX, 1024, 512);
    for (int dir = 0; dir < 2; ++dir)
        gemm(srcX, 512, eWih0b + (size_t)dir * 2048 * 512, 512, eb0 + dir * 2048,
             XW + (size_t)dir * 1024 * 2048, nullptr, 2048, 1024, 2048, 512, 0);
    k_enc_scan<<<2, 1024, 0, stream>>>(eWhh0b, XW, x1f, x1b);
    for (int dir = 0; dir < 2; ++dir)
        gemm(x1b, 1024, eWih1b + (size_t)dir * 2048 * 1024, 1024, eb1 + dir * 2048,
             XW + (size_t)dir * 1024 * 2048, nullptr, 2048, 1024, 2048, 1024, 0);
    k_enc_scan<<<2, 1024, 0, stream>>>(eWhh1b, XW, annf, nullptr);

    // --- decoder beam search ---
    k_dec_init<<<(80 * 512 + 255) / 256, 256, 0, stream>>>(att_s, h0_s, c0_s, h1_s, c1_s,
                                                           token_s, score_s, fin_s);
    for (int t = 0; t < 96; ++t) {
        k_build_in<<<160, 256, 0, stream>>>(token_s, tgt_emb, att_s, h0_s, h1_s, xcat0b, xcat1b);
        gemm(xcat0b, 1536, Wcat0b, 1536, db0, G0, nullptr, 2048, 80, 2048, 1536, 0);
        k_lstm_pw<<<160, 256, 0, stream>>>(G0, c0_s, c0t, h0t, xcat1b, 1024, 0);
        gemm(xcat1b, 1024, Wcat1b, 1024, db1, G1, nullptr, 2048, 80, 2048, 1024, 0);
        k_lstm_pw<<<160, 256, 0, stream>>>(G1, c1_s, c1t, h1t, catab, 1536, 1024);
        // q = h1 @ att_Wa   (A = bf16 h1 stored inside catab at offset 1024, lda=1536)
        gemm(catab + 1024, 1536, WaTb, 512, nullptr, qbuf, nullptr, 1024, 80, 1024, 512, 0);
        k_attn<<<80, 256, 0, stream>>>(qbuf, annf, input_enc, catab, mosta);
        // new_att = tanh([ctx, h1] @ att_Wc^T), keep f32 + bf16 copies
        gemm(catab, 1536, Wcb, 1536, nullptr, nattf, nattb, 512, 80, 512, 1536, 1);
        // logits = new_att @ proj_W^T + proj_b
        gemm(nattb, 512, projWb, 512, pb, logits, nullptr, 32000, 80, 32000, 512, 0);
        k_lse<<<80, 256, 0, stream>>>(logits, rowlse);
        k_topk<<<16, 256, 0, stream>>>(logits, rowlse, score_s, token_s, fin_s, beam_b,
                                       mosta, out, t);
        k_gather<<<160, 256, 0, stream>>>(beam_b, nattf, h0t, c0t, h1t, c1t,
                                          att_s, h0_s, c0_s, h1_s, c1_s);
    }
}
